// AdaptiveRankLSTM8_30365418783340
// MI455X (gfx1250) — compile-verified, matrix operand order checked by reading, not run on hardware
//
#include <hip/hip_runtime.h>
#include <hip/hip_bf16.h>

// ---------------------------------------------------------------------------
// TT-LSTM on MI455X (gfx1250).
//  Step 1: collapse TT cores into dense per-gate 256x256 operators, packed in
//          bf16 WMMA B-fragment layout (build_pack_weights).
//  Step 2: ih_all[t][b][1024] = x @ Wih^T + bias via v_wmma_f32_16x16x32_bf16.
//  Step 3: persistent recurrent kernel: 64 blocks = 4 independent batch
//          groups x 16 blocks. Each block owns one (batch-tile, hid-col-tile)
//          cell tile: its 4 waves compute the 4 gate tiles for that cell tile
//          into LDS, __syncthreads, elementwise update (c in registers), then
//          ONE 16-block barrier per step for packed-h visibility.
// ---------------------------------------------------------------------------

typedef __attribute__((ext_vector_type(16))) __bf16 v16bf;
typedef __attribute__((ext_vector_type(8)))  float  v8f;

#define NGATE 4
#define BATCH 64
#define SEQT  1024
#define DIN   256
#define HID   256
#define NCOL  1024          // NGATE*HID
#define NTILES 64           // NCOL/16
#define KFRAGS 8            // DIN/32
#define FRAG_ELEMS 512      // 32 lanes * 16 bf16
#define RBLOCKS 64          // persistent blocks (4 groups x 16)

// ---- workspace layout (bytes) ----
constexpr size_t OFF_CTR     = 0;                                  // 4 x {cnt,gen} u32, 64B apart
constexpr size_t OFF_BIAS_IH = 256;                                // 1024 f32
constexpr size_t OFF_BIAS_HH = OFF_BIAS_IH + 1024 * 4;             // 1024 f32
constexpr size_t WB_BYTES    = (size_t)NTILES * KFRAGS * FRAG_ELEMS * 2;
constexpr size_t OFF_WB_IH   = OFF_BIAS_HH + 1024 * 4;             // bf16 packed B frags
constexpr size_t OFF_WB_HH   = OFF_WB_IH + WB_BYTES;
constexpr size_t OFF_HA      = OFF_WB_HH + WB_BYTES;               // 4*8*512 bf16 packed h
constexpr size_t OFF_IH      = OFF_HA + 4 * KFRAGS * FRAG_ELEMS * 2; // [T][B][NCOL] f32 (256MB)

// K index inside a 16x32 bf16 A fragment: lane = h8*16 + m, element e = v*2+p
__device__ __forceinline__ int kmap(int e, int h8) {
  int v = e >> 1, p = e & 1;
  return ((v & 3) << 1) + p + (h8 << 3) + ((v >> 2) << 4);
}

__device__ __forceinline__ float sigf(float x) {
  return 1.0f / (1.0f + __expf(-x));
}

// scatter one h value into the packed bf16 A-fragment buffer
__device__ __forceinline__ void store_hA(__bf16* hA, int b, int j, float v) {
  int mt = b >> 4, m = b & 15;
  int kf = j >> 5, k = j & 31;
  int hi = k >> 4, k2 = k & 15, h8 = k2 >> 3, k3 = k2 & 7;
  int vv = hi * 4 + (k3 >> 1), p = k3 & 1;
  int lane = h8 * 16 + m, e = vv * 2 + p;
  hA[(mt * KFRAGS + kf) * FRAG_ELEMS + lane * 16 + e] = (__bf16)v;
}

// barrier among `nblk` blocks sharing one {cnt,gen} pair
__device__ __forceinline__ void group_sync(unsigned* cnt, unsigned* gen, unsigned nblk) {
  __syncthreads();
  if (threadIdx.x == 0) {
    unsigned g = __hip_atomic_load(gen, __ATOMIC_RELAXED, __HIP_MEMORY_SCOPE_AGENT);
    unsigned prev = __hip_atomic_fetch_add(cnt, 1u, __ATOMIC_ACQ_REL, __HIP_MEMORY_SCOPE_AGENT);
    if (prev + 1u == nblk) {
      __hip_atomic_store(cnt, 0u, __ATOMIC_RELAXED, __HIP_MEMORY_SCOPE_AGENT);
      __hip_atomic_fetch_add(gen, 1u, __ATOMIC_RELEASE, __HIP_MEMORY_SCOPE_AGENT);
    } else {
      while (__hip_atomic_load(gen, __ATOMIC_ACQUIRE, __HIP_MEMORY_SCOPE_AGENT) == g)
        __builtin_amdgcn_s_sleep(1);
    }
  }
  __syncthreads();
  __threadfence();   // agent-scope acquire for all lanes
}

// ---------------------------------------------------------------------------
// Kernel 1: collapse TT cores -> dense W (256x256 per gate), pack bf16 B frags
// grid = 8 blocks: blockIdx = which*4 + gate   (which: 0=ih, 1=hh)
// ---------------------------------------------------------------------------
__global__ void __launch_bounds__(256) build_pack_weights(
    const float* ih_g0, const float* ih_g1, const float* ih_g2, const float* ih_g3, const float* ih_b,
    const float* hh_g0, const float* hh_g1, const float* hh_g2, const float* hh_g3, const float* hh_b,
    unsigned char* ws)
{
  const int which = blockIdx.x >> 2;
  const int g     = blockIdx.x & 3;
  const float* c0 = (which ? hh_g0 : ih_g0) + g * 256;   // [16m,16r]
  const float* c1 = (which ? hh_g1 : ih_g1) + g * 4096;  // [16r,16n,16s]
  const float* c2 = (which ? hh_g2 : ih_g2) + g * 4096;  // [16s,16o,16t]
  const float* c3 = (which ? hh_g3 : ih_g3) + g * 256;   // [16t,16p]
  const float* bi = (which ? hh_b  : ih_b ) + g * HID;
  float*  bias_ws = (float*)(ws + (which ? OFF_BIAS_HH : OFF_BIAS_IH)) + g * HID;
  __bf16* wB      = (__bf16*)(ws + (which ? OFF_WB_HH : OFF_WB_IH));

  __shared__ float sA[4096];   // g1 then g2
  __shared__ float sT[4096];   // T1[m,n,s]
  __shared__ float sU[4096];   // U[s,o,p]
  __shared__ float s0[256];    // g0
  __shared__ float s3[256];    // g3

  s0[threadIdx.x] = c0[threadIdx.x];
  for (int i = threadIdx.x; i < 4096; i += 256) sA[i] = c1[i];
  __syncthreads();
  // T1[m,n,s] = sum_r g0[m,r] * g1[r,n,s]
  for (int i = threadIdx.x; i < 4096; i += 256) {
    int m = i >> 8, n = (i >> 4) & 15, s = i & 15;
    float acc = 0.f;
    #pragma unroll
    for (int r = 0; r < 16; ++r) acc += s0[m * 16 + r] * sA[r * 256 + n * 16 + s];
    sT[i] = acc;
  }
  __syncthreads();
  s3[threadIdx.x] = c3[threadIdx.x];
  for (int i = threadIdx.x; i < 4096; i += 256) sA[i] = c2[i];
  __syncthreads();
  // U[s,o,p] = sum_t g2[s,o,t] * g3[t,p]
  for (int i = threadIdx.x; i < 4096; i += 256) {
    int s = i >> 8, o = (i >> 4) & 15, p = i & 15;
    float acc = 0.f;
    #pragma unroll
    for (int t = 0; t < 16; ++t) acc += sA[s * 256 + o * 16 + t] * s3[t * 16 + p];
    sU[i] = acc;
  }
  __syncthreads();
  // W[out=(o,p), in=(m,n)] = sum_s T1[m,n,s]*U[s,o,p]; pack into B fragments
  for (int idx = threadIdx.x; idx < 65536; idx += 256) {
    int ntl = idx >> 12;
    int r   = idx & 4095;
    int kf  = r >> 9;
    int r2  = r & 511;
    int lane = r2 >> 4;
    int e    = r2 & 15;
    int ncol = lane & 15, h8 = lane >> 4;
    int k    = kf * 32 + kmap(e, h8);
    int orow = ntl * 16 + ncol;
    int m = k >> 4, n = k & 15, o = orow >> 4, p = orow & 15;
    float acc = 0.f;
    #pragma unroll
    for (int s = 0; s < 16; ++s) acc += sT[m * 256 + n * 16 + s] * sU[s * 256 + o * 16 + p];
    wB[(size_t)((g * 16 + ntl) * KFRAGS + kf) * FRAG_ELEMS + lane * 16 + e] = (__bf16)acc;
  }
  bias_ws[threadIdx.x] = bi[threadIdx.x];
}

// ---------------------------------------------------------------------------
// Kernel 2: zero packed h0 and the 4 barrier pairs
// ---------------------------------------------------------------------------
__global__ void init_state(unsigned char* ws) {
  int gid = blockIdx.x * blockDim.x + threadIdx.x;
  __bf16* hA = (__bf16*)(ws + OFF_HA);
  if (gid < 4 * KFRAGS * FRAG_ELEMS) hA[gid] = (__bf16)0.f;
  if (gid < 4) {
    unsigned* c = (unsigned*)(ws + OFF_CTR) + gid * 16;  // 64B stride
    c[0] = 0u; c[1] = 0u;
  }
}

// ---------------------------------------------------------------------------
// Kernel 3: ih_all[t][b][1024] = x @ Wih^T + bias  (bf16 WMMA)
// grid = 4096 M-tiles * 8 N-groups; 256 threads = 8 waves, 1 N-tile per wave
// ---------------------------------------------------------------------------
__global__ void __launch_bounds__(256) ih_gemm(const float* __restrict__ x,
                                               unsigned char* ws) {
  const int mt = blockIdx.x >> 3;
  const int ngrp = blockIdx.x & 7;
  const int w = threadIdx.x >> 5;
  const int lane = threadIdx.x & 31;
  const int nt = ngrp * 8 + w;
  const __bf16* wB  = (const __bf16*)(ws + OFF_WB_IH);
  const float* bias = (const float*)(ws + OFF_BIAS_IH);
  float* ihall      = (float*)(ws + OFF_IH);

  // stage 16x256 fp32 A panel -> bf16 WMMA A-fragment layout in LDS
  __shared__ __align__(32) __bf16 ldsA[KFRAGS * FRAG_ELEMS];
  for (int idx = threadIdx.x; idx < KFRAGS * FRAG_ELEMS; idx += 256) {
    int kf = idx >> 9, r = idx & 511, ln = r >> 4, e = r & 15;
    int m = ln & 15, h8 = ln >> 4;
    int k = kf * 32 + kmap(e, h8);
    ldsA[idx] = (__bf16)x[(size_t)(mt * 16 + m) * DIN + k];
  }
  __syncthreads();

  const int n = lane & 15, h8 = lane >> 4;
  const int col = nt * 16 + n;
  v8f acc;
  const float b = bias[col];
  #pragma unroll
  for (int v = 0; v < 8; ++v) acc[v] = b;
  #pragma unroll
  for (int kf = 0; kf < KFRAGS; ++kf) {
    v16bf a  = *(const v16bf*)&ldsA[kf * FRAG_ELEMS + lane * 16];
    v16bf bb = *(const v16bf*)&wB[(size_t)(nt * KFRAGS + kf) * FRAG_ELEMS + lane * 16];
    acc = __builtin_amdgcn_wmma_f32_16x16x32_bf16(false, a, false, bb, (short)0, acc,
                                                  false, false);
  }
  // GEMM row r = b*T + t  ->  store at [t][b][col] for streaming reads later
  #pragma unroll
  for (int v = 0; v < 8; ++v) {
    int r = mt * 16 + v + h8 * 8;
    int bb_ = r >> 10;          // r / SEQT
    int tt  = r & 1023;         // r % SEQT
    ihall[((size_t)tt * BATCH + bb_) * NCOL + col] = acc[v];
  }
}

// ---------------------------------------------------------------------------
// Kernel 4: persistent recurrent kernel. 64 blocks x 128 threads (4 waves).
// blockIdx = mt*16 + ct. Wave w computes gate tile g=w at column tile ct for
// batch tile mt into LDS; __syncthreads; elementwise update of the block's
// own 16x16 cell tile (c in registers); one 16-block barrier per step.
// ---------------------------------------------------------------------------
__global__ void __launch_bounds__(128) lstm_recur(unsigned char* ws,
                                                  float* __restrict__ out) {
  const int mt = blockIdx.x >> 4;                         // batch group 0..3
  const int ct = blockIdx.x & 15;                         // hid column tile
  unsigned* cnt = (unsigned*)(ws + OFF_CTR) + mt * 16;    // 64B-strided pairs
  unsigned* gen = cnt + 1;
  const __bf16* wB   = (const __bf16*)(ws + OFF_WB_HH);
  const float*  hhb  = (const float*)(ws + OFF_BIAS_HH);
  __bf16* hA         = (__bf16*)(ws + OFF_HA);
  const float* ihall = (const float*)(ws + OFF_IH);

  __shared__ __align__(16) float sgates[NGATE * 256];     // 4 gate tiles, 4KB

  const int lane = threadIdx.x & 31;
  const int g    = threadIdx.x >> 5;                      // gate = wave id
  const int nt   = g * 16 + ct;                           // weight N tile
  const int n = lane & 15, h8 = lane >> 4;
  const int col = nt * 16 + n;                            // = g*256 + ct*16 + n
  const float bb = hhb[col];

  // phase-B cell assignment: 2 cells per thread within the 16x16 tile
  const int idx0 = threadIdx.x * 2, idx1 = idx0 + 1;
  const int lb0 = idx0 >> 4, lj0 = idx0 & 15;
  const int lb1 = idx1 >> 4, lj1 = idx1 & 15;
  const int b0 = mt * 16 + lb0, j0 = ct * 16 + lj0;
  const int b1 = mt * 16 + lb1, j1 = ct * 16 + lj1;
  float c0v = 0.f, c1v = 0.f, h0v = 0.f, h1v = 0.f;

  for (int t = 0; t < SEQT; ++t) {
    // ---- phase A: one 16x16 gate tile per wave (gates stay in LDS) -----
    v8f acc;
    #pragma unroll
    for (int v = 0; v < 8; ++v)
      acc[v] = ihall[((size_t)t * BATCH + mt * 16 + v + h8 * 8) * NCOL + col] + bb;
    #pragma unroll
    for (int kf = 0; kf < KFRAGS; ++kf) {
      v16bf a = *(const v16bf*)&hA[(mt * KFRAGS + kf) * FRAG_ELEMS + lane * 16];
      v16bf w = *(const v16bf*)&wB[(size_t)(nt * KFRAGS + kf) * FRAG_ELEMS + lane * 16];
      acc = __builtin_amdgcn_wmma_f32_16x16x32_bf16(false, a, false, w, (short)0, acc,
                                                    false, false);
    }
    #pragma unroll
    for (int v = 0; v < 8; ++v)
      sgates[g * 256 + (v + h8 * 8) * 16 + n] = acc[v];

    // hide next step's ih DRAM latency behind this step's tail
    if (t + 1 < SEQT) {
      const float* nxt =
          &ihall[((size_t)(t + 1) * BATCH + mt * 16 + (lane & 15)) * NCOL + nt * 16];
      __builtin_prefetch(nxt, 0, 3);
    }

    __syncthreads();

    // ---- phase B: elementwise LSTM update of the block's own cells -----
    {
      float gi = sigf (sgates[0 * 256 + lb0 * 16 + lj0]);
      float gf = sigf (sgates[1 * 256 + lb0 * 16 + lj0]);
      float gg = tanhf(sgates[2 * 256 + lb0 * 16 + lj0]);
      float go = sigf (sgates[3 * 256 + lb0 * 16 + lj0]);
      c0v = gf * c0v + gi * gg;
      h0v = go * tanhf(c0v);
      out[((size_t)b0 * SEQT + t) * HID + j0] = h0v;
      store_hA(hA, b0, j0, h0v);
    }
    {
      float gi = sigf (sgates[0 * 256 + lb1 * 16 + lj1]);
      float gf = sigf (sgates[1 * 256 + lb1 * 16 + lj1]);
      float gg = tanhf(sgates[2 * 256 + lb1 * 16 + lj1]);
      float go = sigf (sgates[3 * 256 + lb1 * 16 + lj1]);
      c1v = gf * c1v + gi * gg;
      h1v = go * tanhf(c1v);
      out[((size_t)b1 * SEQT + t) * HID + j1] = h1v;
      store_hA(hA, b1, j1, h1v);
    }

    // one barrier per step: packed-h visibility within this batch group
    group_sync(cnt, gen, 16u);
  }

  // final (h, c) appended after the [B,T,H] output
  const size_t base = (size_t)BATCH * SEQT * HID;
  out[base + b0 * HID + j0] = h0v;
  out[base + b1 * HID + j1] = h1v;
  out[base + BATCH * HID + b0 * HID + j0] = c0v;
  out[base + BATCH * HID + b1 * HID + j1] = c1v;
}

// ---------------------------------------------------------------------------
extern "C" void kernel_launch(void* const* d_in, const int* in_sizes, int n_in,
                              void* d_out, int out_size, void* d_ws, size_t ws_size,
                              hipStream_t stream) {
  (void)in_sizes; (void)n_in; (void)out_size; (void)ws_size;
  const float* x     = (const float*)d_in[0];
  const float* ih_g0 = (const float*)d_in[1];
  const float* ih_g1 = (const float*)d_in[2];
  const float* ih_g2 = (const float*)d_in[3];
  const float* ih_g3 = (const float*)d_in[4];
  const float* ih_b  = (const float*)d_in[5];
  const float* hh_g0 = (const float*)d_in[6];
  const float* hh_g1 = (const float*)d_in[7];
  const float* hh_g2 = (const float*)d_in[8];
  const float* hh_g3 = (const float*)d_in[9];
  const float* hh_b  = (const float*)d_in[10];
  unsigned char* ws  = (unsigned char*)d_ws;
  float* out         = (float*)d_out;

  build_pack_weights<<<8, 256, 0, stream>>>(ih_g0, ih_g1, ih_g2, ih_g3, ih_b,
                                            hh_g0, hh_g1, hh_g2, hh_g3, hh_b, ws);
  init_state<<<64, 256, 0, stream>>>(ws);
  ih_gemm<<<(BATCH * SEQT / 16) * 8, 256, 0, stream>>>(x, ws);
  lstm_recur<<<RBLOCKS, 128, 0, stream>>>(ws, out);
}
